// MMD_19189913878688
// MI455X (gfx1250) — matrix operand order repaired and checked by reading, not compiled, not run on hardware
//
#include <hip/hip_runtime.h>
#include <hip/hip_bf16.h>

// ---------------------------------------------------------------------------
// MMD(source, target) with exp(-gamma * cdist) kernel, gamma = 1.
// N = M = 8192, D = 256. Fully fused: bf16 WMMA GEMM tiles + norms + sqrt +
// exp + reduction; only a scalar partial sum per block ever touches HBM.
// ---------------------------------------------------------------------------

typedef __attribute__((ext_vector_type(16))) __bf16   v16bf;
typedef __attribute__((ext_vector_type(8)))  float    v8f;
typedef __attribute__((ext_vector_type(4)))  unsigned int u32x4;

union FragAB {
    u32x4 q[2];   // two 16-byte chunks (16 bf16 total)
    v16bf v;
};

#define NROWS 8192
#define DDIM  256

// ---------------------------------------------------------------------------
// Pre-pass: one wave (32 lanes) per row. Converts fp32 -> bf16 and computes
// the row squared-norm with a wave shuffle reduction.
// ---------------------------------------------------------------------------
__global__ __launch_bounds__(256)
void convert_rows(const float* __restrict__ in,
                  __bf16* __restrict__ out,
                  float* __restrict__ norms)
{
    const int row  = blockIdx.x * (blockDim.x >> 5) + (threadIdx.x >> 5);
    const int lane = threadIdx.x & 31;
    if (row >= NROWS) return;

    const float* __restrict__ rp = in  + (size_t)row * DDIM;
    __bf16*      __restrict__ op = out + (size_t)row * DDIM;

    float s = 0.0f;
#pragma unroll
    for (int i = 0; i < DDIM / 32; ++i) {
        float v = rp[lane + i * 32];
        s += v * v;
        op[lane + i * 32] = (__bf16)v;
    }
#pragma unroll
    for (int off = 16; off > 0; off >>= 1)
        s += __shfl_xor(s, off, 32);
    if (lane == 0) norms[row] = s;
}

__global__ void zero_out(float* __restrict__ out)
{
    if (threadIdx.x == 0 && blockIdx.x == 0) out[0] = 0.0f;
}

// ---------------------------------------------------------------------------
// Fused tile kernel: C_tile = A * B^T via v_wmma_f32_16x16x32_bf16, epilogue
// sum(exp(-sqrt(relu(na + nb - 2c)))) reduced to one atomicAdd per block.
//
// Block = 256 threads = 8 waves. Block tile 128(M) x 128(N); wave tile
// 32(M) x 64(N) = 2 x 4 accumulators of 16x16. K = 256, unrolled in 8 steps
// of 32. bf16 operands live entirely in the 192MB L2 (8 MB total).
// ---------------------------------------------------------------------------
__global__ __launch_bounds__(256, 2)
void mmd_tile(const __bf16* __restrict__ A,
              const __bf16* __restrict__ B,
              const float*  __restrict__ na,
              const float*  __restrict__ nb,
              float weight,
              float* __restrict__ out)
{
    const int lane  = threadIdx.x & 31;
    const int wave  = threadIdx.x >> 5;
    const int waveM = wave & 3;          // 4 waves along M
    const int waveN = wave >> 2;         // 2 waves along N
    const int mBase = blockIdx.y * 128 + waveM * 32;
    const int nBase = blockIdx.x * 128 + waveN * 64;
    const int r     = lane & 15;         // row/col within 16
    const int h     = lane >> 4;         // half-wave select

    v8f zero = {};
    v8f acc[2][4];
#pragma unroll
    for (int mi = 0; mi < 2; ++mi)
#pragma unroll
        for (int ni = 0; ni < 4; ++ni)
            acc[mi][ni] = zero;

#pragma unroll
    for (int kk = 0; kk < DDIM / 32; ++kk) {
        const int k0 = kk * 32;

        // A fragment (16x32 bf16): lane holds row M = r, K chunks
        // [h*8 .. h*8+7] and [h*8+16 .. h*8+23]  (ISA 16-bit A layout).
        FragAB a[2];
#pragma unroll
        for (int mi = 0; mi < 2; ++mi) {
            const __bf16* p = A + (size_t)(mBase + mi * 16 + r) * DDIM + k0 + h * 8;
            a[mi].q[0] = *(const u32x4*)(p);
            a[mi].q[1] = *(const u32x4*)(p + 16);
        }

        // B fragment (32x16 bf16): lane holds column N = r (== row of B since
        // C = A*B^T), K chunks [h*8 .. h*8+7] and [h*8+16 .. h*8+23].
        FragAB b[4];
#pragma unroll
        for (int ni = 0; ni < 4; ++ni) {
            const __bf16* p = B + (size_t)(nBase + ni * 16 + r) * DDIM + k0 + h * 8;
            b[ni].q[0] = *(const u32x4*)(p);
            b[ni].q[1] = *(const u32x4*)(p + 16);
        }

#pragma unroll
        for (int mi = 0; mi < 2; ++mi)
#pragma unroll
            for (int ni = 0; ni < 4; ++ni)
                acc[mi][ni] = __builtin_amdgcn_wmma_f32_16x16x32_bf16(
                    /*neg_a=*/false, a[mi].v,
                    /*neg_b=*/false, b[ni].v,
                    /*c_mod=*/(short)0, acc[mi][ni],
                    /*reuse_a=*/false, /*reuse_b=*/false);
    }

    // Epilogue. C layout: VGPR rr, lanes 0-15 -> M = rr, N = lane;
    //                                lanes 16-31 -> M = rr + 8, N = lane - 16.
    float sum = 0.0f;
#pragma unroll
    for (int mi = 0; mi < 2; ++mi) {
        float nav[8];
#pragma unroll
        for (int rr = 0; rr < 8; ++rr)
            nav[rr] = na[mBase + mi * 16 + h * 8 + rr];
#pragma unroll
        for (int ni = 0; ni < 4; ++ni) {
            const float nbv = nb[nBase + ni * 16 + r];
#pragma unroll
            for (int rr = 0; rr < 8; ++rr) {
                float sq = nav[rr] + nbv - 2.0f * acc[mi][ni][rr];
                sq = fmaxf(sq, 0.0f);
                sum += __expf(-__builtin_sqrtf(sq));
            }
        }
    }

    // Wave reduction, then block reduction through LDS, one atomic per block.
#pragma unroll
    for (int off = 16; off > 0; off >>= 1)
        sum += __shfl_xor(sum, off, 32);

    __shared__ float wsum[8];
    if (lane == 0) wsum[wave] = sum;
    __syncthreads();
    if (threadIdx.x == 0) {
        float t = 0.0f;
#pragma unroll
        for (int i = 0; i < 8; ++i) t += wsum[i];
        atomicAdd(out, t * weight);
    }
}

// ---------------------------------------------------------------------------
extern "C" void kernel_launch(void* const* d_in, const int* in_sizes, int n_in,
                              void* d_out, int out_size, void* d_ws, size_t ws_size,
                              hipStream_t stream)
{
    (void)in_sizes; (void)n_in; (void)out_size; (void)ws_size;

    const float* src = (const float*)d_in[0];  // [8192, 256] fp32
    const float* tgt = (const float*)d_in[1];  // [8192, 256] fp32
    float* out = (float*)d_out;                // scalar fp32

    char* ws = (char*)d_ws;
    const size_t NB = (size_t)NROWS * DDIM * sizeof(__bf16);   // 4 MB each
    __bf16* Sbf = (__bf16*)(ws);
    __bf16* Tbf = (__bf16*)(ws + NB);
    float*  ns  = (float*)(ws + 2 * NB);
    float*  nt  = (float*)(ws + 2 * NB + NROWS * sizeof(float));

    zero_out<<<1, 64, 0, stream>>>(out);

    // 8 rows per block (8 waves), 1024 blocks per matrix.
    convert_rows<<<NROWS / 8, 256, 0, stream>>>(src, Sbf, ns);
    convert_rows<<<NROWS / 8, 256, 0, stream>>>(tgt, Tbf, nt);

    const float invNM = 1.0f / ((float)NROWS * (float)NROWS);
    dim3 grid(NROWS / 128, NROWS / 128);

    // mean(k_ss) + mean(k_tt) - 2*mean(k_st)
    mmd_tile<<<grid, 256, 0, stream>>>(Sbf, Sbf, ns, ns,         invNM, out);
    mmd_tile<<<grid, 256, 0, stream>>>(Tbf, Tbf, nt, nt,         invNM, out);
    mmd_tile<<<grid, 256, 0, stream>>>(Sbf, Tbf, ns, nt, -2.0f * invNM, out);
}
